// HolographicMemory_50457275793839
// MI455X (gfx1250) — matrix operand order repaired and checked by reading, not compiled, not run on hardware
//
#include <hip/hip_runtime.h>

#define DIM   1024
#define MEM   512
#define HEAD  512
#define NTOT  32768
#define BM    64          // query rows per workgroup
#define WAVES 4
#define TPB   (WAVES * 32)

#define PANEL_HALFS (16 * DIM)                       // 16384 halfs = 32 KB (max panel)
#define LDS_SB      (BM * DIM * 2)                   // Xb:  [BM][DIM] bf16 = 128 KB
#define LDS_TIL     (LDS_SB + BM * MEM * 2)          // Sb:  [BM][MEM] bf16 =  64 KB
#define LDS_TOTAL   (LDS_TIL + 2 * PANEL_HALFS * 2)  // 2 staging buffers   =  64 KB -> 256 KB

typedef __attribute__((ext_vector_type(16))) __bf16 v16bf;
typedef __attribute__((ext_vector_type(2)))  __bf16 v2bf;
typedef __attribute__((ext_vector_type(8)))  float  v8f;
union Frag { v16bf v; uint4 q[2]; };

// Native f32<->bf16 conversions (v_cvt_pk_bf16_f32 on gfx1250, RNE)
__device__ __forceinline__ unsigned short f2bf(float f) {
  return __builtin_bit_cast(unsigned short, (__bf16)f);
}
__device__ __forceinline__ float bf2f(unsigned short b) {
  return (float)__builtin_bit_cast(__bf16, b);
}
__device__ __forceinline__ unsigned int pack2bf(float a, float b) {
  v2bf p; p.x = (__bf16)a; p.y = (__bf16)b;
  return __builtin_bit_cast(unsigned int, p);
}

// ---- CDNA5 async global->LDS copy (ASYNCcnt), with sync fallback ----------
#if __has_builtin(__builtin_amdgcn_global_load_async_to_lds_b128)
#define HAVE_ASYNC 1
typedef int v4i_ __attribute__((__vector_size__(16)));
typedef __attribute__((address_space(1))) v4i_ gv4i_t;   // global 16B vector
typedef __attribute__((address_space(3))) v4i_ lv4i_t;   // LDS    16B vector
#endif

__device__ __forceinline__ void cp16(void* l, const void* g) {
#ifdef HAVE_ASYNC
  __builtin_amdgcn_global_load_async_to_lds_b128(
      (gv4i_t*)(unsigned long long)g,
      (lv4i_t*)(unsigned int)(unsigned long long)l, 0, 0);
#else
  *(uint4*)l = *(const uint4*)g;
#endif
}

__device__ __forceinline__ void wait_async() {
#ifdef HAVE_ASYNC
#if __has_builtin(__builtin_amdgcn_s_wait_asynccnt)
  __builtin_amdgcn_s_wait_asynccnt(0);
#else
  asm volatile("s_wait_asynccnt 0x0" ::: "memory");
#endif
#endif
}

__device__ __forceinline__ void issue_panel(unsigned short* dstL, const unsigned short* srcG,
                                            int n16, int tid) {
  uint4* d = (uint4*)dstL;
  const uint4* s = (const uint4*)srcG;
  for (int i = tid; i < n16; i += TPB) cp16(&d[i], &s[i]);
}

// Pack KV into L2-resident bf16 operand layouts:
//   Kc[m][h]  = concat(keys_real, keys_imag)[m][h]      (512 x 1024, row-major)
//   Vt[h][m]  = concat(values_real, values_imag)[m][h]  (transposed: 1024 x 512)
__global__ void holo_pack(const float* __restrict__ kr, const float* __restrict__ ki,
                          const float* __restrict__ vr, const float* __restrict__ vi,
                          unsigned short* __restrict__ Kc, unsigned short* __restrict__ Vt) {
  int idx = blockIdx.x * blockDim.x + threadIdx.x;   // MEM*DIM threads
  int m = idx >> 10;
  int h = idx & (DIM - 1);
  float kv = (h < HEAD) ? kr[m * HEAD + h] : ki[m * HEAD + (h - HEAD)];
  float vv = (h < HEAD) ? vr[m * HEAD + h] : vi[m * HEAD + (h - HEAD)];
  Kc[m * DIM + h] = f2bf(kv);
  Vt[h * MEM + m] = f2bf(vv);
}

__global__ __launch_bounds__(TPB, 1)
void holo_main(const float* __restrict__ x, const unsigned short* __restrict__ Kc,
               const unsigned short* __restrict__ Vt, float* __restrict__ out) {
  extern __shared__ unsigned char smem[];
  unsigned short* Xb = (unsigned short*)smem;                // [BM][DIM] bf16
  unsigned short* Sb = (unsigned short*)(smem + LDS_SB);     // [BM][MEM] bf16 energies -> probs
  unsigned short* T0 = (unsigned short*)(smem + LDS_TIL);    // staging buf 0 (32 KB)
  unsigned short* T1 = T0 + PANEL_HALFS;                     // staging buf 1 (32 KB)

  const int tid   = threadIdx.x;
  const int lane  = tid & 31;
  const int wave  = tid >> 5;
  const int lidx  = lane & 15;    // N index (B/C/D) or M index (A)
  const int lhalf = lane >> 4;    // half-wave selector per ISA layouts
  const int mrow  = wave * 16;    // this wave's first row within the block
  const long row0 = (long)blockIdx.x * BM;
  const float temp = 0.04419417382415922f;   // HEAD^-0.5

  // ---- Stage X block -> bf16 LDS (vectorized f32x4 reads, 8B LDS writes)
  {
    const float4* xs = (const float4*)(x + row0 * DIM);
    for (int i = tid; i < BM * DIM / 4; i += TPB) {
      float4 v = xs[i];
      *(uint2*)(&Xb[i * 4]) = make_uint2(pack2bf(v.x, v.y), pack2bf(v.z, v.w));
    }
  }
  // prologue: async-DMA K panel 0 into T0 (overlaps with X conversion finishing)
  issue_panel(T0, Kc, PANEL_HALFS / 8, tid);
  __syncthreads();

  // ---- GEMM1: S = Xb(64x1024) @ Kc^T(1024x512); double-buffered K panels
  for (int nt = 0; nt < MEM / 16; ++nt) {
    wait_async();                 // my panel-nt DMA done
    __syncthreads();              // everyone's DMA done; everyone past panel nt-1
    unsigned short* cur = (nt & 1) ? T1 : T0;
    if (nt + 1 < MEM / 16)        // overlap next panel DMA with this tile's WMMAs
      issue_panel((nt & 1) ? T0 : T1, Kc + (size_t)(nt + 1) * PANEL_HALFS,
                  PANEL_HALFS / 8, tid);
    v8f acc = {0.f, 0.f, 0.f, 0.f, 0.f, 0.f, 0.f, 0.f};
    #pragma unroll 4
    for (int kt = 0; kt < DIM / 32; ++kt) {
      const int k0 = kt * 32;
      Frag A, B;
      // A 16x32 bf16: lane m=lidx; regs0-3 K=k0+8*lhalf..+7, regs4-7 K=+16
      const unsigned short* ap = &Xb[(mrow + lidx) * DIM + k0 + lhalf * 8];
      A.q[0] = *(const uint4*)ap;
      A.q[1] = *(const uint4*)(ap + 16);
      // B 32x16 bf16: lane n=lidx; regs0-3 K=k0+8*lhalf..+7, regs4-7 K=+16
      const unsigned short* bp = &cur[lidx * DIM + k0 + lhalf * 8];
      B.q[0] = *(const uint4*)bp;
      B.q[1] = *(const uint4*)(bp + 16);
      acc = __builtin_amdgcn_wmma_f32_16x16x32_bf16(false, A.v, false, B.v,
                                                    (short)0, acc, false, false);
    }
    const int n0 = nt * 16;
    #pragma unroll
    for (int r = 0; r < 8; ++r)   // D row r+8*lhalf, col lidx; fold temperature here
      Sb[(mrow + r + 8 * lhalf) * MEM + n0 + lidx] = f2bf(acc[r] * temp);
  }

  // prefetch GEMM2 panel 0 (Vt rows 0..15) into T0 while softmax runs
  // (T0 free: every wave passed the nt=31 barrier, which followed the last T0 read)
  issue_panel(T0, Vt, 16 * MEM / 8, tid);

  // ---- Softmax over 512 cols, per-wave (each wave owns its 16 rows), in place
  for (int r = 0; r < 16; ++r) {
    unsigned short* srow = &Sb[(mrow + r) * MEM];
    float v[16], mx = -3.0e38f;
    #pragma unroll
    for (int i = 0; i < 16; ++i) { v[i] = bf2f(srow[lane + 32 * i]); mx = fmaxf(mx, v[i]); }
    #pragma unroll
    for (int off = 16; off >= 1; off >>= 1) mx = fmaxf(mx, __shfl_xor(mx, off, 32));
    float sum = 0.f;
    #pragma unroll
    for (int i = 0; i < 16; ++i) { v[i] = __expf(v[i] - mx); sum += v[i]; }
    #pragma unroll
    for (int off = 16; off >= 1; off >>= 1) sum += __shfl_xor(sum, off, 32);
    const float inv = 1.0f / sum;
    #pragma unroll
    for (int i = 0; i < 16; ++i) srow[lane + 32 * i] = f2bf(v[i] * inv);
  }

  // ---- GEMM2: O(64x1024) = P(64x512) @ [Vr|Vi](512x1024); double-buffered V panels
  const unsigned short* Pw = &Sb[mrow * MEM];   // this wave's [16][512] bf16 probs
  for (int nt = 0; nt < DIM / 16; ++nt) {
    wait_async();
    __syncthreads();
    unsigned short* cur = (nt & 1) ? T1 : T0;
    if (nt + 1 < DIM / 16)
      issue_panel((nt & 1) ? T0 : T1, Vt + (size_t)(nt + 1) * 16 * MEM,
                  16 * MEM / 8, tid);
    v8f acc = {0.f, 0.f, 0.f, 0.f, 0.f, 0.f, 0.f, 0.f};
    #pragma unroll 4
    for (int kt = 0; kt < MEM / 32; ++kt) {
      const int k0 = kt * 32;
      Frag A, B;
      const unsigned short* ap = &Pw[lidx * MEM + k0 + lhalf * 8];
      A.q[0] = *(const uint4*)ap;
      A.q[1] = *(const uint4*)(ap + 16);
      const unsigned short* bp = &cur[lidx * MEM + k0 + lhalf * 8];
      B.q[0] = *(const uint4*)bp;
      B.q[1] = *(const uint4*)(bp + 16);
      acc = __builtin_amdgcn_wmma_f32_16x16x32_bf16(false, A.v, false, B.v,
                                                    (short)0, acc, false, false);
    }
    const int n0 = nt * 16;
    float* op = out + (row0 + mrow) * DIM + n0;
    #pragma unroll
    for (int r = 0; r < 8; ++r)
      op[(size_t)(r + 8 * lhalf) * DIM + lidx] = acc[r];
  }
}

extern "C" void kernel_launch(void* const* d_in, const int* in_sizes, int n_in,
                              void* d_out, int out_size, void* d_ws, size_t ws_size,
                              hipStream_t stream) {
  const float* x  = (const float*)d_in[0];
  const float* kr = (const float*)d_in[1];
  const float* ki = (const float*)d_in[2];
  const float* vr = (const float*)d_in[3];
  const float* vi = (const float*)d_in[4];
  float* out = (float*)d_out;

  unsigned short* Kc = (unsigned short*)d_ws;                  // 1 MB
  unsigned short* Vt = Kc + (size_t)MEM * DIM;                 // 1 MB

  holo_pack<<<(MEM * DIM) / 256, 256, 0, stream>>>(kr, ki, vr, vi, Kc, Vt);

  (void)hipFuncSetAttribute(reinterpret_cast<const void*>(holo_main),
                            hipFuncAttributeMaxDynamicSharedMemorySize, LDS_TOTAL);
  holo_main<<<NTOT / BM, TPB, LDS_TOTAL, stream>>>(x, Kc, Vt, out);
}
